// RetNet_24077586662057
// MI455X (gfx1250) — matrix operand order repaired and checked by reading, not compiled, not run on hardware
//
#include <hip/hip_runtime.h>

// ---------------------------------------------------------------------------
// RetNet block stack for MI455X (gfx1250, wave32, WMMA bf16)
// B=4, S=2048, D=1024, L=4
// ---------------------------------------------------------------------------

#define BQ 4
#define SQ 2048
#define DQ 1024
#define LQ 4
#define BS (BQ * SQ)          // 8192 rows
#define LDW 40                // LDS tile row stride in halfwords (80B: 16B-aligned,
                              // bank-conflict-free for b128 across 16 lanes)

// Async global->LDS path (gfx1250 GLOBAL_LOAD_ASYNC_TO_LDS_B128), if toolchain
// exposes the builtins; otherwise fall back to synchronous staging.
#ifdef __has_builtin
#if __has_builtin(__builtin_amdgcn_global_load_async_to_lds_b128) && \
    __has_builtin(__builtin_amdgcn_s_wait_asynccnt)
#define HAVE_ASYNC_LDS 1
#endif
#endif
#ifndef HAVE_ASYNC_LDS
#define HAVE_ASYNC_LDS 0
#endif

typedef __attribute__((ext_vector_type(16))) __bf16 v16bf;
typedef __attribute__((ext_vector_type(8)))  float  v8f;
typedef __attribute__((__vector_size__(4 * sizeof(int)))) int v4i;

#if HAVE_ASYNC_LDS
typedef __attribute__((address_space(1))) v4i* gas_v4i;
typedef __attribute__((address_space(3))) v4i* las_v4i;
#endif

union FragBF { unsigned int u[8]; uint4 u4[2]; v16bf v; };
union FragF  { float f[8];        v8f  v; };

__device__ __forceinline__ unsigned short f2bf(float f) {
    unsigned int u = __float_as_uint(f);
    u += 0x7FFFu + ((u >> 16) & 1u);          // round-to-nearest-even
    return (unsigned short)(u >> 16);
}

__device__ __forceinline__ v8f wmma_bf16(v16bf a, v16bf b, v8f c) {
    // D = A(16x32) * B(32x16) + C, f32 accumulate
    return __builtin_amdgcn_wmma_f32_16x16x32_bf16(
        /*neg_a=*/false, a, /*neg_b=*/false, b,
        /*c_mod=*/(short)0, c, /*reuse_a=*/false, /*reuse_b=*/false);
}

// ---------------------------------------------------------------------------
// f32 -> bf16 conversion
// ---------------------------------------------------------------------------
__global__ __launch_bounds__(256) void k_cvt(const float* __restrict__ in,
                                             unsigned short* __restrict__ out,
                                             long n) {
    long i = (long)blockIdx.x * 256 + threadIdx.x;
    if (i < n) out[i] = f2bf(in[i]);
}

// ---------------------------------------------------------------------------
// Generic bf16 GEMM: C[M,N] = A[M,K] * Bw[K,N] (+bias), WMMA tiled.
// Block tile 128x128, BK=32, 8 waves (4x2), each wave 32x64 (2x4 WMMA tiles).
// mode 0: f32 out; mode 1: bf16 out; mode 2: gelu(exact) -> bf16 out.
// M % 128 == 0, N % 128 == 0, K % 32 == 0 (always true here).
// ---------------------------------------------------------------------------
__global__ __launch_bounds__(256, 1) void k_gemm(const unsigned short* __restrict__ A,
                                                 const unsigned short* __restrict__ Bw,
                                                 const float* __restrict__ bias,
                                                 void* __restrict__ out,
                                                 int M, int N, int K, int mode) {
    __shared__ __attribute__((aligned(16))) unsigned short Asm[128][LDW]; // [m][k]
    __shared__ __attribute__((aligned(16))) unsigned short Bsm[128][LDW]; // [n][k] (transposed)

    const int tid = threadIdx.x;
    const int w = tid >> 5, lane = tid & 31;
    const int lane16 = lane & 15, laneHi = lane >> 4;
    const int waveM = w & 3, waveN = w >> 2;          // 4 x 2 wave grid
    const long m0 = (long)blockIdx.y * 128;
    const long n0 = (long)blockIdx.x * 128;

    FragF acc[2][4];
    #pragma unroll
    for (int mi = 0; mi < 2; ++mi)
        #pragma unroll
        for (int ni = 0; ni < 4; ++ni)
            #pragma unroll
            for (int r = 0; r < 8; ++r) acc[mi][ni].f[r] = 0.f;

    for (int k0 = 0; k0 < K; k0 += 32) {
        // stage A tile 128x32 (row-major)
        {
            int row = tid >> 1;
            int c4 = (tid & 1) * 2;                   // uint4 index (0 or 2)
            const unsigned short* gp = A + (size_t)(m0 + row) * K + k0 + c4 * 8;
            unsigned short* lp = &Asm[row][c4 * 8];
#if HAVE_ASYNC_LDS
            __builtin_amdgcn_global_load_async_to_lds_b128(
                (gas_v4i)gp, (las_v4i)lp, 0, 0);
            __builtin_amdgcn_global_load_async_to_lds_b128(
                (gas_v4i)(gp + 8), (las_v4i)(lp + 8), 0, 0);
#else
            const uint4* g = (const uint4*)gp;
            uint4* sh = (uint4*)lp;
            sh[0] = g[0];
            sh[1] = g[1];
#endif
        }
        // stage B tile 32x128, transposed into Bsm[n][k] (element transpose ->
        // must round-trip through VGPRs)
        {
            int kk = tid >> 3;
            int nstart = (tid & 7) * 16;
            const uint4* g = (const uint4*)(Bw + (size_t)(k0 + kk) * N + n0 + nstart);
            uint4 d0 = g[0], d1 = g[1];
            unsigned int dw[8] = {d0.x, d0.y, d0.z, d0.w, d1.x, d1.y, d1.z, d1.w};
            #pragma unroll
            for (int q = 0; q < 8; ++q) {
                int n = nstart + 2 * q;
                Bsm[n][kk]     = (unsigned short)(dw[q] & 0xFFFFu);
                Bsm[n + 1][kk] = (unsigned short)(dw[q] >> 16);
            }
        }
        // prefetch next K-tile into L2 while this one computes
        if (k0 + 32 < K) {
            __builtin_prefetch(A + (size_t)(m0 + (tid >> 1)) * K + k0 + 32, 0, 1);
            __builtin_prefetch(Bw + (size_t)(k0 + 32 + (tid >> 3)) * N + n0 + (tid & 7) * 16, 0, 1);
        }
#if HAVE_ASYNC_LDS
        __builtin_amdgcn_s_wait_asynccnt(0);
#endif
        __syncthreads();

        // A fragments (2 live), B fragments one at a time (low VGPR pressure)
        FragBF af[2];
        #pragma unroll
        for (int mi = 0; mi < 2; ++mi) {
            const uint4* sh = (const uint4*)&Asm[waveM * 32 + mi * 16 + lane16][0];
            af[mi].u4[0] = sh[laneHi];
            af[mi].u4[1] = sh[2 + laneHi];
        }
        #pragma unroll
        for (int ni = 0; ni < 4; ++ni) {
            const uint4* sh = (const uint4*)&Bsm[waveN * 64 + ni * 16 + lane16][0];
            FragBF bfr;
            bfr.u4[0] = sh[laneHi];
            bfr.u4[1] = sh[2 + laneHi];
            acc[0][ni].v = wmma_bf16(af[0].v, bfr.v, acc[0][ni].v);
            acc[1][ni].v = wmma_bf16(af[1].v, bfr.v, acc[1][ni].v);
        }
        __syncthreads();
    }

    // epilogue
    #pragma unroll
    for (int mi = 0; mi < 2; ++mi)
        #pragma unroll
        for (int ni = 0; ni < 4; ++ni) {
            long col = n0 + waveN * 64 + ni * 16 + lane16;
            float bv = bias ? bias[col] : 0.f;
            long rbase = m0 + waveM * 32 + mi * 16 + 8 * laneHi;
            #pragma unroll
            for (int r = 0; r < 8; ++r) {
                long row = rbase + r;
                float v = acc[mi][ni].f[r] + bv;
                if (mode == 0) {
                    ((float*)out)[row * N + col] = v;
                } else {
                    if (mode == 2) v = 0.5f * v * (1.f + erff(v * 0.70710678118f));
                    ((unsigned short*)out)[row * N + col] = f2bf(v);
                }
            }
        }
}

// ---------------------------------------------------------------------------
// Retention: flash-style causal pass over kqv (bf16 [B,S,3D]: k|q|v).
// One block per (16-row tile, batch); 8 waves, each owns a 128-wide D slice.
// out (bf16 [B,S,D]) = (tril(q k^T)/32/(1+sqrt(row)) / clip(|rowsum|,1)) @ v
// ---------------------------------------------------------------------------
__global__ __launch_bounds__(256, 1) void k_retention(const unsigned short* __restrict__ kqv,
                                                      unsigned short* __restrict__ outb) {
    extern __shared__ __attribute__((aligned(16))) char smem[];
    unsigned short* vT = (unsigned short*)smem;                 // [1024][LDW] (n-major)
    float* s_tile = (float*)(smem + 1024 * LDW * 2);            // [16][32]
    float* rs     = s_tile + 16 * 32;                           // [16]

    const int itile = blockIdx.x, b = blockIdx.y;
    const int tid = threadIdx.x;
    const int w = tid >> 5, lane = tid & 31;
    const int lane16 = lane & 15, laneHi = lane >> 4;
    const int row0 = itile * 16;
    const int RS = 3 * DQ;                                      // 3072

    // preload q fragments for this wave's 128-wide D slice (4 K-steps of 32)
    FragBF qf[4];
    {
        const uint4* qrow =
            (const uint4*)(kqv + (size_t)(b * SQ + row0 + lane16) * RS + DQ);
        #pragma unroll
        for (int ks = 0; ks < 4; ++ks) {
            qf[ks].u4[0] = qrow[w * 16 + ks * 4 + laneHi];
            qf[ks].u4[1] = qrow[w * 16 + ks * 4 + 2 + laneHi];
        }
    }

    FragF acc[8];
    #pragma unroll
    for (int ni = 0; ni < 8; ++ni)
        #pragma unroll
        for (int r = 0; r < 8; ++r) acc[ni].f[r] = 0.f;
    if (tid < 16) rs[tid] = 0.f;

    const int jjmax = (row0 + 15) >> 5;
    for (int jj = 0; jj <= jjmax; ++jj) {
        const int jc0 = jj * 32;
        __syncthreads();                                        // prev iter reads done
        // zero score tile
        s_tile[tid] = 0.f;
        s_tile[tid + 256] = 0.f;
        // stage v tile [32 rows][1024 cols] transposed into vT[n][k]
        {
            int kk = tid >> 3;
            int c4 = (tid & 7) * 16;                            // uint4 start in row
            const uint4* vrow =
                (const uint4*)(kqv + (size_t)(b * SQ + jc0 + kk) * RS + 2 * DQ);
            for (int u = 0; u < 16; ++u) {
                uint4 d = vrow[c4 + u];
                int dcol = 8 * (c4 + u);
                unsigned int dw[4] = {d.x, d.y, d.z, d.w};
                #pragma unroll
                for (int j = 0; j < 4; ++j) {
                    vT[(dcol + 2 * j) * LDW + kk]     = (unsigned short)(dw[j] & 0xFFFFu);
                    vT[(dcol + 2 * j + 1) * LDW + kk] = (unsigned short)(dw[j] >> 16);
                }
            }
        }
        __syncthreads();

        // partial q k^T over this wave's D slice -> atomic reduce into s_tile
        #pragma unroll
        for (int nt = 0; nt < 2; ++nt) {
            FragF c;
            #pragma unroll
            for (int r = 0; r < 8; ++r) c.f[r] = 0.f;
            const uint4* krow =
                (const uint4*)(kqv + (size_t)(b * SQ + jc0 + nt * 16 + lane16) * RS);
            #pragma unroll
            for (int ks = 0; ks < 4; ++ks) {
                FragBF kb;
                kb.u4[0] = krow[w * 16 + ks * 4 + laneHi];
                kb.u4[1] = krow[w * 16 + ks * 4 + 2 + laneHi];
                c.v = wmma_bf16(qf[ks].v, kb.v, c.v);
            }
            #pragma unroll
            for (int r = 0; r < 8; ++r)
                atomicAdd(&s_tile[(r + 8 * laneHi) * 32 + nt * 16 + lane16], c.f[r]);
        }
        __syncthreads();

        // wave 0: mask + scale + row-sum
        if (tid < 32) {
            int m = tid >> 1;
            int row = row0 + m;
            float sc = 0.03125f / (1.f + sqrtf((float)row));    // 1/sqrt(1024) * row_norm
            float part = 0.f;
            for (int q = 0; q < 16; ++q) {
                int c = (tid & 1) * 16 + q;
                int col = jc0 + c;
                float v = (col <= row) ? s_tile[m * 32 + c] * sc : 0.f;
                s_tile[m * 32 + c] = v;
                part += v;
            }
            atomicAdd(&rs[m], part);
        }
        __syncthreads();

        // s @ v for this wave's D slice
        FragBF af;
        {
            int m = lane16;
            #pragma unroll
            for (int p = 0; p < 8; ++p) {
                int k0 = (p & 3) * 2 + 8 * (2 * (p >> 2) + laneHi);
                unsigned short l0 = f2bf(s_tile[m * 32 + k0]);
                unsigned short l1 = f2bf(s_tile[m * 32 + k0 + 1]);
                af.u[p] = (unsigned int)l0 | ((unsigned int)l1 << 16);
            }
        }
        #pragma unroll
        for (int ni = 0; ni < 8; ++ni) {
            int n = w * 128 + ni * 16 + lane16;
            const uint4* vr = (const uint4*)(vT + n * LDW);
            FragBF bfr;
            bfr.u4[0] = vr[laneHi];
            bfr.u4[1] = vr[2 + laneHi];
            acc[ni].v = wmma_bf16(af.v, bfr.v, acc[ni].v);
        }
    }
    __syncthreads();

    // divide by clip(|rowsum|,1), store bf16
    #pragma unroll
    for (int ni = 0; ni < 8; ++ni) {
        int dcol = w * 128 + ni * 16 + lane16;
        #pragma unroll
        for (int r = 0; r < 8; ++r) {
            int m = r + 8 * laneHi;
            float denom = fmaxf(fabsf(rs[m]), 1.f);
            outb[(size_t)(b * SQ + row0 + m) * DQ + dcol] = f2bf(acc[ni].f[r] / denom);
        }
    }
}

// ---------------------------------------------------------------------------
// SwiGLU: a * silu(gate) on [rows, 2D] -> bf16 [rows, D]
// ---------------------------------------------------------------------------
__global__ __launch_bounds__(256) void k_swiglu(const float* __restrict__ m1,
                                                unsigned short* __restrict__ outb) {
    long i = (long)blockIdx.x * 256 + threadIdx.x;   // over BS*D
    long row = i >> 10;                              // / 1024
    int d = (int)(i & 1023);
    float a = m1[row * 2048 + d];
    float g = m1[row * 2048 + 1024 + d];
    float s = g / (1.f + expf(-g));
    outb[i] = f2bf(a * s);
}

// ---------------------------------------------------------------------------
// GroupNorm(1,D) stats over (S,D) per batch: partial sum / sumsq -> atomics
// ---------------------------------------------------------------------------
__global__ __launch_bounds__(256) void k_gnstats(const float* __restrict__ x,
                                                 float* __restrict__ stats) {
    __shared__ float s1[256], s2[256];
    const long perBatch = (long)SQ * DQ;             // 2M
    int tid = threadIdx.x;
    long base = (long)blockIdx.y * perBatch + (long)blockIdx.x * 2048 + tid;
    float a = 0.f, bb = 0.f;
    #pragma unroll
    for (int q = 0; q < 8; ++q) {
        float v = x[base + q * 256];
        a += v; bb += v * v;
    }
    s1[tid] = a; s2[tid] = bb;
    __syncthreads();
    for (int st = 128; st > 0; st >>= 1) {
        if (tid < st) { s1[tid] += s1[tid + st]; s2[tid] += s2[tid + st]; }
        __syncthreads();
    }
    if (tid == 0) {
        atomicAdd(&stats[blockIdx.y * 2 + 0], s1[0]);
        atomicAdd(&stats[blockIdx.y * 2 + 1], s2[0]);
    }
}

__global__ __launch_bounds__(256) void k_gnapply(const float* __restrict__ m2,
                                                 const float* __restrict__ h,
                                                 const float* __restrict__ stats,
                                                 const float* __restrict__ g,
                                                 const float* __restrict__ bta,
                                                 float* __restrict__ yf,
                                                 unsigned short* __restrict__ yb) {
    const long perBatch = (long)SQ * DQ;
    long bIdx = blockIdx.y;
    long i = (long)blockIdx.x * 256 + threadIdx.x;
    float mu = stats[bIdx * 2] / (float)perBatch;
    float var = stats[bIdx * 2 + 1] / (float)perBatch - mu * mu;
    float rstd = rsqrtf(var + 1e-5f);
    long idx = bIdx * perBatch + i;
    int d = (int)(i & 1023);
    float v = (m2[idx] - mu) * rstd * g[d] + bta[d] + h[idx];
    yf[idx] = v;
    yb[idx] = f2bf(v);
}

// ---------------------------------------------------------------------------
// LayerNorm over last dim (D=1024): out = LN(f2 + y); writes f32 + bf16
// ---------------------------------------------------------------------------
__global__ __launch_bounds__(256) void k_layernorm(const float* __restrict__ f2,
                                                   const float* __restrict__ y,
                                                   const float* __restrict__ g,
                                                   const float* __restrict__ bta,
                                                   float* __restrict__ outf,
                                                   unsigned short* __restrict__ outb) {
    __shared__ float s1[256], s2[256];
    long row = blockIdx.x;
    int tid = threadIdx.x;
    float vals[4];
    float a = 0.f, bb = 0.f;
    #pragma unroll
    for (int q = 0; q < 4; ++q) {
        long idx = row * DQ + tid + q * 256;
        float v = f2[idx] + y[idx];
        vals[q] = v; a += v; bb += v * v;
    }
    s1[tid] = a; s2[tid] = bb;
    __syncthreads();
    for (int st = 128; st > 0; st >>= 1) {
        if (tid < st) { s1[tid] += s1[tid + st]; s2[tid] += s2[tid + st]; }
        __syncthreads();
    }
    float mu = s1[0] / (float)DQ;
    float var = s2[0] / (float)DQ - mu * mu;
    float rstd = rsqrtf(var + 1e-5f);
    #pragma unroll
    for (int q = 0; q < 4; ++q) {
        int d = tid + q * 256;
        float v = (vals[q] - mu) * rstd * g[d] + bta[d];
        outf[row * DQ + d] = v;
        outb[row * DQ + d] = f2bf(v);
    }
}

// ---------------------------------------------------------------------------
// Host orchestration
// ---------------------------------------------------------------------------
extern "C" void kernel_launch(void* const* d_in, const int* in_sizes, int n_in,
                              void* d_out, int out_size, void* d_ws, size_t ws_size,
                              hipStream_t stream) {
    (void)in_sizes; (void)n_in; (void)out_size; (void)ws_size;

    const float* x_in = (const float*)d_in[0];
    const float* Wkqv = (const float*)d_in[1];
    const float* Wm1  = (const float*)d_in[2];
    const float* bm1  = (const float*)d_in[3];
    const float* Wm2  = (const float*)d_in[4];
    const float* bm2  = (const float*)d_in[5];
    const float* gn_g = (const float*)d_in[6];
    const float* gn_b = (const float*)d_in[7];
    const float* Wf1  = (const float*)d_in[8];
    const float* bf1  = (const float*)d_in[9];
    const float* Wf2  = (const float*)d_in[10];
    const float* bf2  = (const float*)d_in[11];
    const float* ln_g = (const float*)d_in[12];
    const float* ln_b = (const float*)d_in[13];

    // workspace carving
    char* p = (char*)d_ws;
    auto alloc = [&](size_t bytes) -> char* {
        char* r = p;
        p += (bytes + 255) & ~(size_t)255;
        return r;
    };
    const size_t nWkqv = (size_t)LQ * DQ * 3 * DQ;
    const size_t nWm1  = (size_t)LQ * DQ * 2 * DQ;
    const size_t nWm2  = (size_t)LQ * DQ * DQ;
    const size_t nWf1  = (size_t)LQ * DQ * 2 * DQ;
    const size_t nWf2  = (size_t)LQ * 2 * DQ * DQ;
    const size_t nBSD  = (size_t)BS * DQ;            // 8M elems

    unsigned short* WkqvB = (unsigned short*)alloc(nWkqv * 2);
    unsigned short* Wm1B  = (unsigned short*)alloc(nWm1 * 2);
    unsigned short* Wm2B  = (unsigned short*)alloc(nWm2 * 2);
    unsigned short* Wf1B  = (unsigned short*)alloc(nWf1 * 2);
    unsigned short* Wf2B  = (unsigned short*)alloc(nWf2 * 2);
    unsigned short* xb    = (unsigned short*)alloc(nBSD * 2);
    unsigned short* retb  = (unsigned short*)alloc(nBSD * 2);
    unsigned short* sb    = (unsigned short*)alloc(nBSD * 2);
    unsigned short* yb    = (unsigned short*)alloc(nBSD * 2);
    float* m2f = (float*)alloc(nBSD * 4);
    float* yf  = (float*)alloc(nBSD * 4);
    float* xf  = (float*)alloc(nBSD * 4);
    char*  slabA = alloc((size_t)64 * 1024 * 1024);  // kqv(48M) | m1(64M) | f1b(32M)+f2(32M)
    float* stats = (float*)alloc(256);

    unsigned short* kqv = (unsigned short*)slabA;
    float*          m1  = (float*)slabA;
    unsigned short* f1b = (unsigned short*)slabA;
    float*          f2f = (float*)(slabA + (size_t)32 * 1024 * 1024);

    auto cvt = [&](const float* src, unsigned short* dst, size_t n) {
        k_cvt<<<dim3((unsigned)((n + 255) / 256)), 256, 0, stream>>>(src, dst, (long)n);
    };
    cvt(Wkqv, WkqvB, nWkqv);
    cvt(Wm1,  Wm1B,  nWm1);
    cvt(Wm2,  Wm2B,  nWm2);
    cvt(Wf1,  Wf1B,  nWf1);
    cvt(Wf2,  Wf2B,  nWf2);
    cvt(x_in, xb,    nBSD);

    const size_t retLds = (size_t)1024 * LDW * 2 + 16 * 32 * 4 + 16 * 4; // ~84KB

    for (int l = 0; l < LQ; ++l) {
        const unsigned short* wk  = WkqvB + (size_t)l * DQ * 3 * DQ;
        const unsigned short* w1  = Wm1B  + (size_t)l * DQ * 2 * DQ;
        const unsigned short* w2  = Wm2B  + (size_t)l * DQ * DQ;
        const unsigned short* wf1 = Wf1B  + (size_t)l * DQ * 2 * DQ;
        const unsigned short* wf2 = Wf2B  + (size_t)l * 2 * DQ * DQ;
        const float* bm1_l = bm1 + (size_t)l * 2 * DQ;
        const float* bm2_l = bm2 + (size_t)l * DQ;
        const float* gg_l  = gn_g + (size_t)l * DQ;
        const float* gb_l  = gn_b + (size_t)l * DQ;
        const float* bf1_l = bf1 + (size_t)l * 2 * DQ;
        const float* bf2_l = bf2 + (size_t)l * DQ;
        const float* lg_l  = ln_g + (size_t)l * DQ;
        const float* lb_l  = ln_b + (size_t)l * DQ;
        const float* hcur  = (l == 0) ? x_in : xf;

        // 1) kqv = x @ Wkqv  -> bf16 [BS, 3D]
        k_gemm<<<dim3(3 * DQ / 128, BS / 128), 256, 0, stream>>>(
            xb, wk, nullptr, kqv, BS, 3 * DQ, DQ, 1);

        // 2) retention -> retb bf16 [BS, D]
        k_retention<<<dim3(SQ / 16, BQ), 256, retLds, stream>>>(kqv, retb);

        // 3) m1 = ret @ Wm1 + bm1 -> f32 [BS, 2D]
        k_gemm<<<dim3(2 * DQ / 128, BS / 128), 256, 0, stream>>>(
            retb, w1, bm1_l, m1, BS, 2 * DQ, DQ, 0);

        // 4) SwiGLU -> sb bf16 [BS, D]
        k_swiglu<<<dim3((unsigned)(nBSD / 256)), 256, 0, stream>>>(m1, sb);

        // 5) m2 = sb @ Wm2 + bm2 -> f32 [BS, D]
        k_gemm<<<dim3(DQ / 128, BS / 128), 256, 0, stream>>>(
            sb, w2, bm2_l, m2f, BS, DQ, DQ, 0);

        // 6) GroupNorm(1,D) per batch + residual -> yf f32, yb bf16
        (void)hipMemsetAsync(stats, 0, BQ * 2 * sizeof(float), stream);
        k_gnstats<<<dim3(1024, BQ), 256, 0, stream>>>(m2f, stats);
        k_gnapply<<<dim3((unsigned)((SQ * (size_t)DQ) / 256), BQ), 256, 0, stream>>>(
            m2f, hcur, stats, gg_l, gb_l, yf, yb);

        // 7) f1 = gelu(y @ Wf1 + bf1) -> bf16 [BS, 2D]
        k_gemm<<<dim3(2 * DQ / 128, BS / 128), 256, 0, stream>>>(
            yb, wf1, bf1_l, f1b, BS, 2 * DQ, DQ, 2);

        // 8) f2 = f1 @ Wf2 + bf2 -> f32 [BS, D]
        k_gemm<<<dim3(DQ / 128, BS / 128), 256, 0, stream>>>(
            f1b, wf2, bf2_l, f2f, BS, DQ, 2 * DQ, 0);

        // 9) out = LayerNorm(f2 + y) -> next x (f32 + bf16); last layer -> d_out
        float* outf = (l == LQ - 1) ? (float*)d_out : xf;
        k_layernorm<<<dim3(BS), 256, 0, stream>>>(f2f, yf, lg_l, lb_l, outf, xb);
    }
}
